// ProLiF_46703474377283
// MI455X (gfx1250) — matrix operand order repaired and checked by reading, not compiled, not run on hardware
//
#include <hip/hip_runtime.h>

typedef __attribute__((ext_vector_type(16))) _Float16 v16h;
typedef __attribute__((ext_vector_type(8)))  float    v8f;

union H16 { v16h h; uint4 u[2]; };

// HW v_sin_f32 computes sin(2*pi*x): fold omega=30 and the 1/2pi into one scale.
#define KSCL 4.774648292756860f   // 30 / (2*pi)

// Load a 16x32 f16 A-fragment slice for this lane: 8 halves at p, 8 at p+8.
__device__ __forceinline__ v16h load_afrag(const _Float16* p) {
    H16 a;
    a.u[0] = *(const uint4*)(p);
    a.u[1] = *(const uint4*)(p + 8);
    return a.h;
}

// Build two 32x16 f16 B-fragments (K-chunks q=0,1) from four 16x16 f32 D tiles.
// D tile t: lane l holds rows (r + 8*(l>>4)) of features [16t,16t+16), col l&15.
// B chunk q: lane l needs features 32q + 16*(l>>4) + j (j=0..15) at col l&15.
// Rows the lane is missing live in lane l^16 at the same column -> one shfl_xor.
__device__ __forceinline__ void build_bfrag(const v8f acc[4], int hl, H16 bfr[2]) {
#pragma unroll
    for (int q = 0; q < 2; ++q) {
#pragma unroll
        for (int r = 0; r < 8; ++r) {
            float own = hl ? acc[2*q + 1][r] : acc[2*q][r];     // tile this lane needs
            float snd = hl ? acc[2*q][r]     : acc[2*q + 1][r]; // tile the partner needs
            float oth = __shfl_xor(snd, 16, 32);                // partner's rows of my tile
            bfr[q].h[r]     = (_Float16)(hl ? oth : own);       // feature offsets 0..7
            bfr[q].h[r + 8] = (_Float16)(hl ? own : oth);       // feature offsets 8..15
        }
    }
}

// r = sin(2*pi * (K*c + biasK))  with biasK pre-scaled by K in LDS.
// One v_fma + one v_sin per element.
__device__ __forceinline__ v8f bias_sin(v8f c, const float* bb) {
    float4 q0 = *(const float4*)(bb);
    float4 q1 = *(const float4*)(bb + 4);
    v8f r;
    r[0] = __builtin_amdgcn_sinf(__builtin_fmaf(KSCL, c[0], q0.x));
    r[1] = __builtin_amdgcn_sinf(__builtin_fmaf(KSCL, c[1], q0.y));
    r[2] = __builtin_amdgcn_sinf(__builtin_fmaf(KSCL, c[2], q0.z));
    r[3] = __builtin_amdgcn_sinf(__builtin_fmaf(KSCL, c[3], q0.w));
    r[4] = __builtin_amdgcn_sinf(__builtin_fmaf(KSCL, c[4], q1.x));
    r[5] = __builtin_amdgcn_sinf(__builtin_fmaf(KSCL, c[5], q1.y));
    r[6] = __builtin_amdgcn_sinf(__builtin_fmaf(KSCL, c[6], q1.z));
    r[7] = __builtin_amdgcn_sinf(__builtin_fmaf(KSCL, c[7], q1.w));
    return r;
}

__global__ __launch_bounds__(256)
void prolif_siren_kernel(const float* __restrict__ emb,
                         const float* __restrict__ W0,
                         const float* __restrict__ b0,
                         const float* __restrict__ Wmid,
                         const float* __restrict__ bmid,
                         const float* __restrict__ Wlast,
                         const float* __restrict__ blast,
                         float* __restrict__ out)
{
    constexpr int B = 65536, N = 16, C = 32, H = 64, O = 4, MID = 4;
    // f16 weights: W0 [0,2048) | Wmid[l] at 2048+l*4096 | Wlast(16x64 padded) at 18432
    __shared__ __attribute__((aligned(16))) _Float16 sW[19456];
    // f32 biases (pre-scaled by KSCL): b0 [0,64) | bmid [64,320) | blast(raw) [320,324)
    __shared__ __attribute__((aligned(16))) float    sB[324];

    const int n    = blockIdx.y;
    const int tid  = threadIdx.x;
    const int lane = tid & 31;
    const int wv   = tid >> 5;
    const int hl   = lane >> 4;   // half-wave index (0/1)
    const int col  = lane & 15;

    // ---------------- preamble: convert this sub-field's weights to f16 in LDS
    {
        const float* g0 = W0 + n * H * C;
        for (int i = tid; i < H * C; i += 256) sW[i] = (_Float16)g0[i];
#pragma unroll
        for (int l = 0; l < MID; ++l) {
            const float* g = Wmid + (size_t)(l * N + n) * H * H;
            _Float16*    d = sW + 2048 + l * H * H;
            for (int i = tid; i < H * H; i += 256) d[i] = (_Float16)g[i];
        }
        const float* gl = Wlast + n * O * H;
        _Float16*    dl = sW + 18432;
        for (int i = tid; i < 16 * H; i += 256) {
            float v = 0.0f;
            if (i < O * H) v = gl[i];
            dl[i] = (_Float16)v;              // rows 4..15 zero-padded
        }
        for (int i = tid; i < H; i += 256) sB[i] = KSCL * b0[n * H + i];
        for (int i = tid; i < MID * H; i += 256) {
            int l = i >> 6, r = i & 63;
            sB[64 + i] = KSCL * bmid[(size_t)(l * N + n) * H + r];
        }
        if (tid < O) sB[320 + tid] = blast[n * O + tid];
    }
    __syncthreads();

    const int tiles   = B / 16;               // 4096 row-tiles
    const int tstride = gridDim.x * 8;        // waves covering the batch

#pragma clang loop unroll(disable)
    for (int t = blockIdx.x * 8 + wv; t < tiles; t += tstride) {
        const int row0 = t * 16;

        // Opaque zero: makes all LDS addresses loop-variant so LICM cannot
        // hoist the (loop-invariant) weight/bias fragments into registers,
        // which previously blew past 256 VGPRs and spilled to scratch.
        int zoff = 0;
        asm volatile("" : "+v"(zoff));
        const _Float16* wlds = sW + zoff;
        const float*    blds = sB + zoff;

        // Prefetch next tile's emb rows into cache (speculative; OOB dropped).
        {
            const float* pf = emb + (size_t)(row0 + tstride * 16 + col) * (N * C) + n * C;
            __builtin_prefetch(pf, 0, 1);
        }

        // ---------------- layer 0: B fragment (x^T) straight from global emb
        H16 bf0;
        {
            const float* src = emb + (size_t)(row0 + col) * (N * C) + n * C + hl * 16;
            float4 p0 = ((const float4*)src)[0];
            float4 p1 = ((const float4*)src)[1];
            float4 p2 = ((const float4*)src)[2];
            float4 p3 = ((const float4*)src)[3];
            bf0.h[0]  = (_Float16)p0.x; bf0.h[1]  = (_Float16)p0.y;
            bf0.h[2]  = (_Float16)p0.z; bf0.h[3]  = (_Float16)p0.w;
            bf0.h[4]  = (_Float16)p1.x; bf0.h[5]  = (_Float16)p1.y;
            bf0.h[6]  = (_Float16)p1.z; bf0.h[7]  = (_Float16)p1.w;
            bf0.h[8]  = (_Float16)p2.x; bf0.h[9]  = (_Float16)p2.y;
            bf0.h[10] = (_Float16)p2.z; bf0.h[11] = (_Float16)p2.w;
            bf0.h[12] = (_Float16)p3.x; bf0.h[13] = (_Float16)p3.y;
            bf0.h[14] = (_Float16)p3.z; bf0.h[15] = (_Float16)p3.w;
        }

        v8f acc[4];
#pragma unroll
        for (int ft = 0; ft < 4; ++ft) {
            v16h a = load_afrag(wlds + (ft * 16 + col) * C + hl * 16);
            v8f  c = {0.f, 0.f, 0.f, 0.f, 0.f, 0.f, 0.f, 0.f};
            c = __builtin_amdgcn_wmma_f32_16x16x32_f16(false, a, false, bf0.h,
                                                       (short)0, c, false, false);
            acc[ft] = bias_sin(c, blds + ft * 16 + hl * 8);
        }

        // ---------------- 4 middle layers (register-resident activations)
#pragma unroll
        for (int l = 0; l < MID; ++l) {
            const _Float16* wbase = wlds + 2048 + l * H * H;
            const float*    bbase = blds + 64 + l * 64;

            H16 bfr[2];
            build_bfrag(acc, hl, bfr);

            v8f nacc[4];
#pragma unroll
            for (int ft = 0; ft < 4; ++ft) {
                v8f c = {0.f, 0.f, 0.f, 0.f, 0.f, 0.f, 0.f, 0.f};
#pragma unroll
                for (int q = 0; q < 2; ++q) {
                    v16h a = load_afrag(wbase + (ft * 16 + col) * H + q * 32 + hl * 16);
                    c = __builtin_amdgcn_wmma_f32_16x16x32_f16(false, a, false, bfr[q].h,
                                                               (short)0, c, false, false);
                }
                nacc[ft] = bias_sin(c, bbase + ft * 16 + hl * 8);
            }
#pragma unroll
            for (int ft = 0; ft < 4; ++ft) acc[ft] = nacc[ft];
        }

        // ---------------- last layer: O=4 (rows 0..3 of one padded 16-row tile)
        {
            H16 bfr[2];
            build_bfrag(acc, hl, bfr);
            v8f c = {0.f, 0.f, 0.f, 0.f, 0.f, 0.f, 0.f, 0.f};
#pragma unroll
            for (int q = 0; q < 2; ++q) {
                v16h a = load_afrag(wlds + 18432 + col * H + q * 32 + hl * 16);
                c = __builtin_amdgcn_wmma_f32_16x16x32_f16(false, a, false, bfr[q].h,
                                                           (short)0, c, false, false);
            }
            if (hl == 0) {   // lanes 0..15 hold output rows o=0..3 in c[0..3]
                float4 bl = *(const float4*)(blds + 320);
                float4 r;
                r.x = c[0] + bl.x;
                r.y = c[1] + bl.y;
                r.z = c[2] + bl.z;
                r.w = c[3] + bl.w;
                *(float4*)(out + (size_t)(row0 + col) * (N * O) + n * O) = r;
            }
        }
    }
}

extern "C" void kernel_launch(void* const* d_in, const int* in_sizes, int n_in,
                              void* d_out, int out_size, void* d_ws, size_t ws_size,
                              hipStream_t stream) {
    (void)in_sizes; (void)n_in; (void)out_size; (void)d_ws; (void)ws_size;
    const float* emb   = (const float*)d_in[0];
    const float* W0    = (const float*)d_in[1];
    const float* b0    = (const float*)d_in[2];
    const float* Wmid  = (const float*)d_in[3];
    const float* bmid  = (const float*)d_in[4];
    const float* Wlast = (const float*)d_in[5];
    const float* blast = (const float*)d_in[6];
    float* out = (float*)d_out;

    dim3 grid(64, 16, 1);   // 64 batch-chunks x 16 sub-fields; 8 waves/block
    dim3 block(256, 1, 1);
    prolif_siren_kernel<<<grid, block, 0, stream>>>(emb, W0, b0, Wmid, bmid,
                                                    Wlast, blast, out);
}
